// MultiheadfAttention_50603304682119
// MI455X (gfx1250) — matrix operand order repaired and checked by reading, not compiled, not run on hardware
//
#include <hip/hip_runtime.h>
#include <hip/hip_bf16.h>
#include <math.h>

// Multi-head attention with top-64 sparse softmax, CDNA5 (gfx1250) wave32 WMMA.
// B=4, T=1024, C=1024, H=16, hd=64, TOPK=64.

#define BATCH 4
#define TSEQ  1024
#define CDIM  1024
#define NHEAD 16
#define HD    64
#define TOPK  64

typedef __attribute__((ext_vector_type(16))) __bf16 bf16x16;
typedef __attribute__((ext_vector_type(8)))  __bf16 bf16x8;
typedef __attribute__((ext_vector_type(8)))  float  f32x8;

static __device__ __forceinline__ f32x8 wmma_bf16(bf16x16 a, bf16x16 b, f32x8 c) {
  // D = A(16x32 bf16) x B(32x16 bf16) + C(16x16 f32)
  return __builtin_amdgcn_wmma_f32_16x16x32_bf16(false, a, false, b, (short)0, c,
                                                 false, false);
}

static __device__ __forceinline__ bf16x16 pack16(bf16x8 lo, bf16x8 hi) {
  return __builtin_shufflevector(lo, hi, 0, 1, 2, 3, 4, 5, 6, 7,
                                 8, 9, 10, 11, 12, 13, 14, 15);
}

// A fragment: 16(M) x 32(K) bf16, src row-major with leading dim ld.
// Lanes 0-15: M=lane, K = 0..7 and 16..23.  Lanes 16-31: M=lane-16, K = 8..15 and 24..31.
static __device__ __forceinline__ bf16x16 load_a_frag(const __bf16* __restrict__ src,
                                                      int ld, int lane) {
  int m  = lane & 15;
  int kb = (lane & 16) ? 8 : 0;
  const __bf16* p = src + (size_t)m * ld + kb;
  bf16x8 lo = *reinterpret_cast<const bf16x8*>(p);        // K = kb .. kb+7
  bf16x8 hi = *reinterpret_cast<const bf16x8*>(p + 16);   // K = kb+16 .. kb+23
  return pack16(lo, hi);
}

// B fragment: 32(K) x 16(N) bf16, where src is the *transposed* operand ([N][K] row-major).
// Lane n=lane&15; lanes 0-15 hold K=0..7 and 16..23, lanes 16-31 hold K=8..15 and 24..31.
static __device__ __forceinline__ bf16x16 load_b_frag(const __bf16* __restrict__ src,
                                                      int ld, int lane) {
  int n  = lane & 15;
  int kb = (lane & 16) ? 8 : 0;
  const __bf16* p = src + (size_t)n * ld + kb;
  bf16x8 lo = *reinterpret_cast<const bf16x8*>(p);
  bf16x8 hi = *reinterpret_cast<const bf16x8*>(p + 16);
  return pack16(lo, hi);
}

static __device__ __forceinline__ int wave_sum_i(int v) {
  #pragma unroll
  for (int o = 16; o > 0; o >>= 1) v += __shfl_xor(v, o, 32);
  return v;
}
static __device__ __forceinline__ float wave_sum_f(float v) {
  #pragma unroll
  for (int o = 16; o > 0; o >>= 1) v += __shfl_xor(v, o, 32);
  return v;
}
static __device__ __forceinline__ float wave_max_f(float v) {
  #pragma unroll
  for (int o = 16; o > 0; o >>= 1) v = fmaxf(v, __shfl_xor(v, o, 32));
  return v;
}
// Order-preserving f32 -> u32 map (monotone for all finite values and +-inf).
static __device__ __forceinline__ unsigned ord_f32(float f) {
  unsigned u = __float_as_uint(f);
  return (u & 0x80000000u) ? ~u : (u | 0x80000000u);
}

// ---------------- conversion / transpose pre-passes ----------------

__global__ void cvt_f32_bf16_kernel(const float* __restrict__ src,
                                    __bf16* __restrict__ dst, int n) {
  int i = blockIdx.x * blockDim.x + threadIdx.x;
  if (i < n) dst[i] = (__bf16)src[i];
}

// W[k][n] (f32, CDIM x CDIM) -> Wt[n][k] (bf16)
__global__ __launch_bounds__(256) void transpose_cvt_kernel(const float* __restrict__ W,
                                                            __bf16* __restrict__ Wt) {
  __shared__ float tile[16][17];
  int x = blockIdx.x * 16 + threadIdx.x;   // n
  int y = blockIdx.y * 16 + threadIdx.y;   // k
  tile[threadIdx.y][threadIdx.x] = W[(size_t)y * CDIM + x];
  __syncthreads();
  int nk = blockIdx.y * 16 + threadIdx.x;  // k
  int nn = blockIdx.x * 16 + threadIdx.y;  // n
  Wt[(size_t)nn * CDIM + nk] = (__bf16)tile[threadIdx.x][threadIdx.y];
}

// ---------------- WMMA GEMM: C(4096x1024) = A(bf16) * Bt^T + bias ----------------
// Wave tile 32(M) x 64(N): B fragments reused across two A fragments.
// MODE 0/1: store bf16 [B][H][T][hd]   (Q / K projections)
// MODE 2:   store bf16 [B][H][hd][T]   (V projection, transposed for P@V B-operand)
// MODE 3:   store f32  [B*T][C]        (output projection)
template <int MODE>
__global__ __launch_bounds__(128)
void gemm32_kernel(const __bf16* __restrict__ A, const __bf16* __restrict__ Bt,
                   const float* __restrict__ bias, void* __restrict__ dst) {
  int lane = threadIdx.x & 31;
  // wave id is uniform within the wave: pin it to an SGPR for scalar control flow
  int wid  = __builtin_amdgcn_readfirstlane(blockIdx.x * 4 + (threadIdx.x >> 5));
  int rt = wid >> 4;        // row tile (32 rows), 0..127
  int cg = wid & 15;        // col group (64 cols), 0..15
  int n0 = cg * 64;
  const __bf16* A0 = A + (size_t)rt * 32 * CDIM;

  f32x8 acc[2][4] = {};
  for (int k0 = 0; k0 < CDIM; k0 += 32) {
    bf16x16 a0 = load_a_frag(A0 + k0, CDIM, lane);
    bf16x16 a1 = load_a_frag(A0 + (size_t)16 * CDIM + k0, CDIM, lane);
    #pragma unroll
    for (int j = 0; j < 4; ++j) {
      bf16x16 b = load_b_frag(Bt + (size_t)(n0 + j * 16) * CDIM + k0, CDIM, lane);
      acc[0][j] = wmma_bf16(a0, b, acc[0][j]);
      acc[1][j] = wmma_bf16(a1, b, acc[1][j]);
    }
  }

  int nlo  = lane & 15;
  int moff = (lane >> 4) << 3;   // C/D layout: VGPR v -> M = v + moff
  #pragma unroll
  for (int j = 0; j < 4; ++j) {
    int col = n0 + j * 16 + nlo;
    float bv = bias[col];
    #pragma unroll
    for (int sub = 0; sub < 2; ++sub) {
      #pragma unroll
      for (int v = 0; v < 8; ++v) {
        int grow = rt * 32 + sub * 16 + v + moff;
        float val = acc[sub][j][v] + bv;
        if (MODE == 3) {
          ((float*)dst)[(size_t)grow * CDIM + col] = val;
        } else {
          int b = grow >> 10, t = grow & (TSEQ - 1);
          int h = col >> 6,  d = col & (HD - 1);
          __bf16* o = (__bf16*)dst;
          if (MODE == 2)
            o[(((size_t)(b * NHEAD + h)) * HD + d) * TSEQ + t] = (__bf16)val;
          else
            o[(((size_t)(b * NHEAD + h)) * TSEQ + t) * HD + d] = (__bf16)val;
        }
      }
    }
  }
}

// ---------------- fused attention: scores -> top-64 -> softmax -> P@V ----------------
// 4 waves (128 threads) per 16-row tile, 4096 blocks.
//   phase 1: score col-tiles round-robin over waves
//   phase 2: top-64 + softmax, 4 rows per wave
//   phase 3: P@V split along s (K dim) with LDS partial reduction
__global__ __launch_bounds__(128)
void attn_topk_kernel(const __bf16* __restrict__ Qp, const __bf16* __restrict__ Kp,
                      const __bf16* __restrict__ Vt, __bf16* __restrict__ Y) {
  __shared__ float S[16 * TSEQ];      // 64 KB score / prob row buffer
  __shared__ float rowScale[16];      // 1/rowsum per row
  __shared__ float Yp[4][16][HD];     // 16 KB per-wave partial y
  int tid  = threadIdx.x;
  int lane = tid & 31;
  int w    = __builtin_amdgcn_readfirstlane(tid >> 5);  // wave id 0..3, SGPR
  int tile = blockIdx.x;
  int rt = tile & 63;                 // row tile within (b,h)
  int bh = tile >> 6;
  int h = bh & (NHEAD - 1);
  int b = bh >> 4;
  int t0 = rt * 16;

  const __bf16* Qb = Qp + (size_t)bh * TSEQ * HD;
  const __bf16* Kb = Kp + (size_t)bh * TSEQ * HD;
  const __bf16* Vb = Vt + (size_t)bh * HD * TSEQ;

  // Q row-tile fragments (K-dim = head dim, two chunks of 32)
  bf16x16 qa0 = load_a_frag(Qb + (size_t)t0 * HD + 0,  HD, lane);
  bf16x16 qa1 = load_a_frag(Qb + (size_t)t0 * HD + 32, HD, lane);

  int nChunks = (t0 + 16 + 31) >> 5;   // 32-wide s-chunks needed for causal extent
  int nColT   = nChunks * 2;           // 16-wide score col tiles
  int Sw      = nColT * 16;
  const float scale = 0.125f;          // 1/sqrt(64)

  int nlo  = lane & 15;
  int moff = (lane >> 4) << 3;

  // ---- phase 1: scores S[t][s] = scale * q.k, causal-masked to -inf ----
  for (int ct = w; ct < nColT; ct += 4) {
    int s0 = ct * 16;
    if (ct + 4 < nColT) {   // uniform guard -> scalar branch; per-lane prefetch addr
      __builtin_prefetch(Kb + (size_t)(s0 + 64 + nlo) * HD + ((lane & 16) ? 32 : 0), 0, 1);
    }
    // load both B fragments first so the loads clause and the waits are partial
    bf16x16 kb0 = load_b_frag(Kb + (size_t)s0 * HD + 0,  HD, lane);
    bf16x16 kb1 = load_b_frag(Kb + (size_t)s0 * HD + 32, HD, lane);
    f32x8 c = {};
    c = wmma_bf16(qa0, kb0, c);
    c = wmma_bf16(qa1, kb1, c);
    #pragma unroll
    for (int v = 0; v < 8; ++v) {
      int m = v + moff;
      float val = ((s0 + nlo) <= (t0 + m)) ? c[v] * scale : -INFINITY;
      S[m * TSEQ + s0 + nlo] = val;
    }
  }
  __syncthreads();

  // ---- phase 2: per-row top-64 threshold (radix binary search) + softmax in place ----
  #pragma unroll 1
  for (int i = 0; i < 4; ++i) {
    int r = w * 4 + i;
    int t = t0 + r;
    int L = t + 1;                      // valid causal length of this row
    float    rv[32];
    unsigned ru[32];
    #pragma unroll
    for (int j = 0; j < 32; ++j) {
      int s = lane + j * 32;
      float x = (s < Sw) ? S[r * TSEQ + s] : -INFINITY;
      rv[j] = x;
      ru[j] = (s < L) ? ord_f32(x) : 0u;   // 0 marks invalid (finite ords are > 0)
    }
    unsigned thr = 0u;
    if (L > TOPK) {
      unsigned cur = 0u;
      #pragma unroll 1
      for (int bit = 31; bit >= 0; --bit) {
        unsigned cand = cur | (1u << bit);
        int cnt = 0;
        #pragma unroll
        for (int j = 0; j < 32; ++j) cnt += (ru[j] >= cand) ? 1 : 0;
        cnt = wave_sum_i(cnt);
        if (cnt >= TOPK) cur = cand;
      }
      thr = cur;                        // exactly the 64th-largest value's code
    }
    float mx = -INFINITY;
    #pragma unroll
    for (int j = 0; j < 32; ++j) mx = fmaxf(mx, rv[j]);
    mx = wave_max_f(mx);
    float sum = 0.f;
    #pragma unroll
    for (int j = 0; j < 32; ++j) {
      int s = lane + j * 32;
      float e = (ru[j] >= thr) ? __expf(rv[j] - mx) : 0.f;  // -inf entries exp to 0
      sum += e;
      if (s < Sw) S[r * TSEQ + s] = e;
    }
    sum = wave_sum_f(sum);
    if (lane == 0) rowScale[r] = 1.0f / sum;
  }
  __syncthreads();

  // ---- phase 3: y(16 x 64) = P @ V; wave w owns s-chunks w, w+4, ... ----
  float myScale = rowScale[lane & 15];  // A-fragment rows: lane&15
  f32x8 acc[4] = {};
  int m  = lane & 15;
  int kb = (lane & 16) ? 8 : 0;
  for (int c = w; c < nChunks; c += 4) {
    int s0 = c * 32;
    bf16x16 a;
    #pragma unroll
    for (int i = 0; i < 8; ++i) {       // fold 1/rowsum while converting to bf16
      a[i]     = (__bf16)(S[m * TSEQ + s0 + kb + i] * myScale);
      a[i + 8] = (__bf16)(S[m * TSEQ + s0 + kb + 16 + i] * myScale);
    }
    #pragma unroll
    for (int j = 0; j < 4; ++j) {
      bf16x16 bf = load_b_frag(Vb + (size_t)(j * 16) * TSEQ + s0, TSEQ, lane);
      acc[j] = wmma_bf16(a, bf, acc[j]);
    }
  }
  #pragma unroll
  for (int j = 0; j < 4; ++j) {
    #pragma unroll
    for (int v = 0; v < 8; ++v) Yp[w][v + moff][j * 16 + nlo] = acc[j][v];
  }
  __syncthreads();

  // reduce 4 partials and store y as bf16 in [B][T][C] for the final GEMM
  #pragma unroll
  for (int e = 0; e < 8; ++e) {
    int idx = tid * 8 + e;              // 0..1023 over the 16x64 tile
    int m2 = idx >> 6, d2 = idx & 63;
    float s = Yp[0][m2][d2] + Yp[1][m2][d2] + Yp[2][m2][d2] + Yp[3][m2][d2];
    Y[((size_t)(b * TSEQ + t0 + m2)) * CDIM + h * HD + d2] = (__bf16)s;
  }
}

// ---------------- host launch ----------------

extern "C" void kernel_launch(void* const* d_in, const int* in_sizes, int n_in,
                              void* d_out, int out_size, void* d_ws, size_t ws_size,
                              hipStream_t stream) {
  (void)in_sizes; (void)n_in; (void)out_size; (void)ws_size;
  const float* q  = (const float*)d_in[0];
  const float* k  = (const float*)d_in[1];
  const float* v  = (const float*)d_in[2];
  // d_in[3] = tgt_mask (causal, recomputed arithmetically)
  const float* Wq = (const float*)d_in[4];
  const float* Wk = (const float*)d_in[5];
  const float* Wv = (const float*)d_in[6];
  const float* Wo = (const float*)d_in[7];
  const float* bq = (const float*)d_in[8];
  const float* bk = (const float*)d_in[9];
  const float* bv = (const float*)d_in[10];
  const float* bo = (const float*)d_in[11];

  const size_t XN = (size_t)BATCH * TSEQ * CDIM;    // 4M elements
  const size_t WN = (size_t)CDIM * CDIM;            // 1M elements
  __bf16* qbf = (__bf16*)d_ws;
  __bf16* kbf = qbf + XN;
  __bf16* vbf = kbf + XN;
  __bf16* Wqt = vbf + XN;
  __bf16* Wkt = Wqt + WN;
  __bf16* Wvt = Wkt + WN;
  __bf16* Wot = Wvt + WN;
  __bf16* Qp  = Wot + WN;
  __bf16* Kp  = Qp + XN;
  __bf16* Vt  = Kp + XN;
  __bf16* Y   = Vt + XN;   // total ~64 MB of workspace

  // bf16 conversion of activations
  cvt_f32_bf16_kernel<<<(int)(XN / 256), 256, 0, stream>>>(q, qbf, (int)XN);
  cvt_f32_bf16_kernel<<<(int)(XN / 256), 256, 0, stream>>>(k, kbf, (int)XN);
  cvt_f32_bf16_kernel<<<(int)(XN / 256), 256, 0, stream>>>(v, vbf, (int)XN);
  // weight transpose + bf16 ([N][K] so WMMA B-fragments are K-contiguous loads)
  dim3 tg(64, 64), tb(16, 16);
  transpose_cvt_kernel<<<tg, tb, 0, stream>>>(Wq, Wqt);
  transpose_cvt_kernel<<<tg, tb, 0, stream>>>(Wk, Wkt);
  transpose_cvt_kernel<<<tg, tb, 0, stream>>>(Wv, Wvt);
  transpose_cvt_kernel<<<tg, tb, 0, stream>>>(Wo, Wot);

  // projections (2048 wave-tiles of 32x64, 4 waves/block)
  gemm32_kernel<0><<<512, 128, 0, stream>>>(qbf, Wqt, bq, Qp);
  gemm32_kernel<1><<<512, 128, 0, stream>>>(kbf, Wkt, bk, Kp);
  gemm32_kernel<2><<<512, 128, 0, stream>>>(vbf, Wvt, bv, Vt);

  // fused attention with top-64 sparsification
  attn_topk_kernel<<<BATCH * NHEAD * (TSEQ / 16), 128, 0, stream>>>(Qp, Kp, Vt, Y);

  // output projection -> f32
  gemm32_kernel<3><<<512, 128, 0, stream>>>(Y, Wot, bo, (float*)d_out);
}
